// NnHalfKPCuda_36498632081980
// MI455X (gfx1250) — matrix operand order repaired and checked by reading, not compiled, not run on hardware
//
#include <hip/hip_runtime.h>
#include <math.h>

typedef __attribute__((ext_vector_type(16))) _Float16 v16h;
typedef __attribute__((ext_vector_type(8)))  float    v8f;
typedef __attribute__((ext_vector_type(4)))  float    f4;
typedef __attribute__((ext_vector_type(4)))  _Float16 h4;

#define NUM_FFT 640
#define FT_OUT  256
#define MAXF    32
#define PPB     16       // positions per block == WMMA M tile
#define HSTRH   520      // hidden LDS stride in halves (512 + 8): keeps every
                         // 8-half run 16-byte aligned, staggers banks

__device__ __forceinline__ float clamp01(float x) {
  return fminf(fmaxf(x, 0.0f), 1.0f);
}

__global__ __launch_bounds__(256) void nnue_halfkp_fused(
    const float* __restrict__ values,
    const int*   __restrict__ stm,
    const int*   __restrict__ nstm,
    const float* __restrict__ ft_w,
    const float* __restrict__ ft_b,
    const float* __restrict__ fft_w,
    const float* __restrict__ fft_b,
    const float* __restrict__ out_w,
    const float* __restrict__ out_b,
    float*       __restrict__ out,
    int B)
{
  __shared__ int      s_si[PPB][MAXF];   // stm feature index
  __shared__ int      s_sf[PPB][MAXF];   // stm index % 640
  __shared__ int      s_ni[PPB][MAXF];   // nstm feature index
  __shared__ int      s_nf[PPB][MAXF];   // nstm index % 640
  __shared__ float    s_v [PPB][MAXF];   // feature values
  __shared__ _Float16 s_wh[2 * FT_OUT];  // output weights (512), f16
  __shared__ _Float16 s_h [PPB * HSTRH]; // clamped hidden, f16, padded rows

  const int tid  = threadIdx.x;
  const int pos0 = blockIdx.x * PPB;

  // ---- stage indices / values / output weights into LDS (coalesced) ----
  for (int i = tid; i < PPB * MAXF; i += 256) {
    const int p = i >> 5, f = i & 31;
    int si = 0, ni = 0; float v = 0.0f;
    if (pos0 + p < B) {
      si = stm   [(size_t)(pos0 + p) * MAXF + f];
      ni = nstm  [(size_t)(pos0 + p) * MAXF + f];
      v  = values[(size_t)(pos0 + p) * MAXF + f];
    }
    s_si[p][f] = si; s_sf[p][f] = si % NUM_FFT;
    s_ni[p][f] = ni; s_nf[p][f] = ni % NUM_FFT;
    s_v [p][f] = v;
  }
  for (int i = tid; i < 2 * FT_OUT; i += 256) s_wh[i] = (_Float16)out_w[i];
  __syncthreads();

  // ---- phase 1: gather-weighted embedding sums (16 lanes per position) ----
  // lane-in-group l owns dims { 64*q + 4*l + j : q=0..3, j=0..3 } so each
  // float4 gather instruction covers 256 contiguous bytes of a table row.
  const int p = tid >> 4;
  const int l = tid & 15;

  float aS[16], aN[16];
  #pragma unroll
  for (int q = 0; q < 4; ++q) {
    const f4 fb = *(const f4*)(ft_b  + q * 64 + l * 4);
    const f4 gb = *(const f4*)(fft_b + q * 64 + l * 4);
    aS[q*4+0] = aN[q*4+0] = fb.x + gb.x;
    aS[q*4+1] = aN[q*4+1] = fb.y + gb.y;
    aS[q*4+2] = aN[q*4+2] = fb.z + gb.z;
    aS[q*4+3] = aN[q*4+3] = fb.w + gb.w;
  }

  for (int f = 0; f < MAXF; ++f) {
    const float v = s_v[p][f];
    const f4* rS  = (const f4*)(ft_w  + (size_t)s_si[p][f] * FT_OUT);
    const f4* rSf = (const f4*)(fft_w + (size_t)s_sf[p][f] * FT_OUT);
    const f4* rN  = (const f4*)(ft_w  + (size_t)s_ni[p][f] * FT_OUT);
    const f4* rNf = (const f4*)(fft_w + (size_t)s_nf[p][f] * FT_OUT);
    #pragma unroll
    for (int q = 0; q < 4; ++q) {
      const f4 a = rS [q * 16 + l];
      const f4 b = rSf[q * 16 + l];
      const f4 c = rN [q * 16 + l];
      const f4 d = rNf[q * 16 + l];
      aS[q*4+0] = fmaf(v, a.x, fmaf(v, b.x, aS[q*4+0]));
      aS[q*4+1] = fmaf(v, a.y, fmaf(v, b.y, aS[q*4+1]));
      aS[q*4+2] = fmaf(v, a.z, fmaf(v, b.z, aS[q*4+2]));
      aS[q*4+3] = fmaf(v, a.w, fmaf(v, b.w, aS[q*4+3]));
      aN[q*4+0] = fmaf(v, c.x, fmaf(v, d.x, aN[q*4+0]));
      aN[q*4+1] = fmaf(v, c.y, fmaf(v, d.y, aN[q*4+1]));
      aN[q*4+2] = fmaf(v, c.z, fmaf(v, d.z, aN[q*4+2]));
      aN[q*4+3] = fmaf(v, c.w, fmaf(v, d.w, aN[q*4+3]));
    }
  }

  // clamp -> f16 -> LDS (packed 8-byte ds_store_b64 per quad)
  #pragma unroll
  for (int q = 0; q < 4; ++q) {
    const int dim0 = q * 64 + l * 4;
    h4 hs, hn;
    #pragma unroll
    for (int j = 0; j < 4; ++j) {
      hs[j] = (_Float16)clamp01(aS[q*4+j]);
      hn[j] = (_Float16)clamp01(aN[q*4+j]);
    }
    *(h4*)&s_h[p * HSTRH + dim0]          = hs;   // stm half  [0,256)
    *(h4*)&s_h[p * HSTRH + FT_OUT + dim0] = hn;   // nstm half [256,512)
  }
  __syncthreads();

  // ---- phase 2: output layer on wave 0 via V_WMMA_F32_16X16X32_F16 ----
  // D[16,16] chain over 16 K-chunks of 32; out_w occupies column N=0 only.
  // A-fragment lane layout (16-bit A, 16x32): lane<16 holds K {0..7,16..23},
  // lane>=16 holds K {8..15,24..31} => two contiguous 8-half (16B) LDS runs.
  if (tid < 32) {                       // wave-uniform: EXEC all-1s at WMMA
    const int  lane  = tid;
    const int  m     = lane & 15;
    const int  sel   = lane >> 4;
    const bool wlane = (lane == 0) | (lane == 16);  // B column 0 holders
    const _Float16* hrow = &s_h[m * HSTRH];
    const f4 z4 = {0.f, 0.f, 0.f, 0.f};

    v8f acc = {0.f, 0.f, 0.f, 0.f, 0.f, 0.f, 0.f, 0.f};
    for (int c = 0; c < 16; ++c) {
      const int kbase = c * 32;
      union { f4 q[2]; v16h v; } ua, ub;
      // A: two 16-byte half-runs
      ua.q[0] = *(const f4*)&hrow[kbase + sel * 8];
      ua.q[1] = *(const f4*)&hrow[kbase + 16 + sel * 8];
      // B: unconditional two 16-byte weight half-runs, then select (v_cndmask, no EXEC)
      const f4 w0 = *(const f4*)&s_wh[kbase + sel * 8];
      const f4 w1 = *(const f4*)&s_wh[kbase + 16 + sel * 8];
      ub.q[0] = wlane ? w0 : z4;
      ub.q[1] = wlane ? w1 : z4;
      acc = __builtin_amdgcn_wmma_f32_16x16x32_f16(
          /*neg_a=*/false, ua.v, /*neg_b=*/false, ub.v,
          /*c_mod=*/(short)0, acc, /*reuse_a=*/false, /*reuse_b=*/false);
    }

    // D column 0: lane 0 holds M=0..7 in acc[0..7], lane 16 holds M=8..15
    if (wlane) {
      const float ob = out_b[0];
      const int   mb = pos0 + sel * 8;
      #pragma unroll
      for (int r = 0; r < 8; ++r) {
        if (mb + r < B) {
          const float x = acc[r] + ob;
          out[mb + r] = 1.0f / (1.0f + expf(-x));
        }
      }
    }
  }
}

extern "C" void kernel_launch(void* const* d_in, const int* in_sizes, int n_in,
                              void* d_out, int out_size, void* d_ws, size_t ws_size,
                              hipStream_t stream) {
  const float* values = (const float*)d_in[0];
  const int*   stm    = (const int*)  d_in[1];
  const int*   nstm   = (const int*)  d_in[2];
  const float* ft_w   = (const float*)d_in[3];
  const float* ft_b   = (const float*)d_in[4];
  const float* fft_w  = (const float*)d_in[5];
  const float* fft_b  = (const float*)d_in[6];
  const float* out_w  = (const float*)d_in[7];
  const float* out_b  = (const float*)d_in[8];
  float*       out    = (float*)d_out;

  const int B      = in_sizes[0] / MAXF;     // 8192
  const int blocks = (B + PPB - 1) / PPB;    // 512

  nnue_halfkp_fused<<<blocks, 256, 0, stream>>>(
      values, stm, nstm, ft_w, ft_b, fft_w, fft_b, out_w, out_b, out, B);
}
